// OptimizedGreenFormer_73727408603193
// MI455X (gfx1250) — matrix operand order repaired and checked by reading, not compiled, not run on hardware
//
#include <hip/hip_runtime.h>
#include <hip/hip_bf16.h>
#include <math.h>

// ---------------------------------------------------------------------------
// OptimizedGreenFormer for MI455X (gfx1250, wave32, WMMA).
// All matmuls use v_wmma_f32_16x16x32_f16 (f16 in, f32 accumulate).
// ---------------------------------------------------------------------------

typedef _Float16 f16;
typedef __attribute__((ext_vector_type(16))) _Float16 v16h;
typedef __attribute__((ext_vector_type(8)))  _Float16 h8;    // 16B chunk
typedef __attribute__((ext_vector_type(8)))  float    v8f;
typedef __attribute__((ext_vector_type(4)))  float    f4;    // 16B chunk

#define BATCH   2
#define NTOK    4096            // 64*64 tokens
#define DIM     768
#define HEADS   12
#define HD      64
#define HIDDEN  1536
#define MTOT    (BATCH*NTOK)    // 8192 rows
#define HINTRES 1024

enum { MODE_QKV = 0, MODE_PROJ = 1, MODE_FC1 = 2, MODE_FC2 = 3 };

// LDS strides: 16B-aligned chunks AND conflict-free bank rotation.
// LDA = 40 halves = 80B = 20 dwords: bank = 4*(5r mod 16) -> 16 distinct rows.
// LDF = 72 halves = 144B = 36 dwords: bank = 4*(9r mod 16) -> 16 distinct rows.
#define LDA 40
#define LDF 72

// ---------------------------------------------------------------------------
// 0) zero the per-batch edge counters
// ---------------------------------------------------------------------------
__global__ void init_counts_kernel(int* cnt) {
    if (threadIdx.x < BATCH) cnt[threadIdx.x] = 0;
}

// ---------------------------------------------------------------------------
// 1) edge mask: 16x16 area pool of alpha_hint -> threshold, count edges
// ---------------------------------------------------------------------------
__global__ __launch_bounds__(256)
void edge_mask_kernel(const float* __restrict__ ah, float* __restrict__ edgeF,
                      int* __restrict__ cnt) {
    int idx = blockIdx.x * 256 + threadIdx.x;       // 0..8191
    int b   = idx >> 12;
    int tok = idx & 4095;
    int th  = tok >> 6, tw = tok & 63;
    const float* base = ah + (size_t)b * HINTRES * HINTRES
                           + (size_t)th * 16 * HINTRES + (size_t)tw * 16;
    float s = 0.f;
    for (int i = 0; i < 16; ++i) {
        const f4* r = (const f4*)(base + i * HINTRES);
        #pragma unroll
        for (int j = 0; j < 4; ++j) {
            f4 v = r[j];
            s += v.x + v.y + v.z + v.w;
        }
    }
    float mean = s * (1.0f / 256.0f);
    int e = (mean > 0.02f) && (mean < 0.98f);
    edgeF[idx] = (float)e;
    if (e) atomicAdd(&cnt[b], 1);
}

// ---------------------------------------------------------------------------
// 2) finalize: per-batch fallback (<64 edge tokens -> all edge), key bias,
//    easy-token count (clamped to >=1)
// ---------------------------------------------------------------------------
__global__ __launch_bounds__(256)
void edge_finalize_kernel(float* __restrict__ edgeF, float* __restrict__ kbias,
                          const int* __restrict__ cnt, float* __restrict__ ecnt) {
    int idx = blockIdx.x * 256 + threadIdx.x;
    int b   = idx >> 12;
    int c   = cnt[b];
    float e = (c < 64) ? 1.0f : edgeF[idx];
    edgeF[idx] = e;
    kbias[idx] = (e > 0.5f) ? 0.0f : -1.0e9f;
    if ((idx & 4095) == 0) {
        int edgeTot = (c < 64) ? NTOK : c;
        ecnt[b] = fmaxf((float)(NTOK - edgeTot), 1.0f);
    }
}

// ---------------------------------------------------------------------------
// 3) LayerNorm, fused with x -> f16 conversion. One block per token.
// ---------------------------------------------------------------------------
__global__ __launch_bounds__(256)
void layernorm_kernel(const float* __restrict__ x, const float* __restrict__ g,
                      const float* __restrict__ bb, f16* __restrict__ x16,
                      f16* __restrict__ xn16) {
    int tok = blockIdx.x;
    int tid = threadIdx.x;
    const float* xp = x + (size_t)tok * DIM;
    float v0 = xp[tid], v1 = xp[tid + 256], v2 = xp[tid + 512];
    float s  = v0 + v1 + v2;
    float sq = v0 * v0 + v1 * v1 + v2 * v2;
    #pragma unroll
    for (int m = 1; m < 32; m <<= 1) {
        s  += __shfl_xor(s,  m, 32);
        sq += __shfl_xor(sq, m, 32);
    }
    __shared__ float ss[8], ssq[8];
    __shared__ float mu_s, rstd_s;
    int wave = tid >> 5, lane = tid & 31;
    if (lane == 0) { ss[wave] = s; ssq[wave] = sq; }
    __syncthreads();
    if (tid == 0) {
        float S = 0.f, SQ = 0.f;
        #pragma unroll
        for (int i = 0; i < 8; ++i) { S += ss[i]; SQ += ssq[i]; }
        float mu  = S * (1.0f / DIM);
        float var = SQ * (1.0f / DIM) - mu * mu;
        mu_s   = mu;
        rstd_s = rsqrtf(fmaxf(var, 0.f) + 1e-5f);
    }
    __syncthreads();
    float mu = mu_s, rstd = rstd_s;
    size_t base = (size_t)tok * DIM;
    x16[base + tid]       = (f16)v0;
    x16[base + tid + 256] = (f16)v1;
    x16[base + tid + 512] = (f16)v2;
    xn16[base + tid]       = (f16)((v0 - mu) * rstd * g[tid]       + bb[tid]);
    xn16[base + tid + 256] = (f16)((v1 - mu) * rstd * g[tid + 256] + bb[tid + 256]);
    xn16[base + tid + 512] = (f16)((v2 - mu) * rstd * g[tid + 512] + bb[tid + 512]);
}

// ---------------------------------------------------------------------------
// 4) Generic WMMA GEMM: C[M,N] = A[M,K] (f16) @ W[K,N] (f32, cvt on load).
//    Block = 256 threads = 8 waves; tile 256(M) x 64(N), K-step 32.
//    Wave w owns rows [w*32, w*32+32) x 64 cols -> 8 accumulators.
//    Register-staged double buffering: global loads for step k+1 issue
//    before the WMMAs of step k (VMEM latency hides under matrix work).
// ---------------------------------------------------------------------------
__global__ __launch_bounds__(256)
void gemm_wmma_kernel(const f16* __restrict__ A, const float* __restrict__ W,
                      const float* __restrict__ bias, int K, int Nw, int mode,
                      f16* __restrict__ outQ, f16* __restrict__ outK,
                      f16* __restrict__ outV, const float* __restrict__ resid,
                      float* __restrict__ outF, f16* __restrict__ outH) {
    __shared__ __align__(16) f16 As[2][256 * LDA];
    __shared__ __align__(16) f16 Bt[2][64 * LDA];   // transposed: Bt[n][k]
    const int tid  = threadIdx.x;
    const int wave = tid >> 5, lane = tid & 31;
    const int mBase = blockIdx.y * 256;
    const int nBase = blockIdx.x * 64;
    const int bk = tid >> 3, bn0 = (tid & 7) * 8;   // W staging coords

    v8f acc[2][4] = {};
    h8 aR[4];       // one full 32-wide A row per thread
    f4 wR[2];       // 8 f32 of W per thread

    // ---- stage K-step 0 ----
    {
        const h8* sa = (const h8*)(A + (size_t)(mBase + tid) * K);
        #pragma unroll
        for (int i = 0; i < 4; ++i) aR[i] = sa[i];
        const f4* sw = (const f4*)(W + (size_t)bk * Nw + nBase + bn0);
        wR[0] = sw[0]; wR[1] = sw[1];
        h8* da = (h8*)&As[0][tid * LDA];
        #pragma unroll
        for (int i = 0; i < 4; ++i) da[i] = aR[i];
        #pragma unroll
        for (int j = 0; j < 4; ++j) Bt[0][(bn0 + j) * LDA + bk]     = (f16)wR[0][j];
        #pragma unroll
        for (int j = 0; j < 4; ++j) Bt[0][(bn0 + 4 + j) * LDA + bk] = (f16)wR[1][j];
    }
    __syncthreads();

    int cur = 0;
    for (int kk = 0; kk < K; kk += 32) {
        const bool more = (kk + 32) < K;
        if (more) {   // issue next tile's global loads before computing
            const h8* sa = (const h8*)(A + (size_t)(mBase + tid) * K + kk + 32);
            #pragma unroll
            for (int i = 0; i < 4; ++i) aR[i] = sa[i];
            const f4* sw = (const f4*)(W + (size_t)(kk + 32 + bk) * Nw + nBase + bn0);
            wR[0] = sw[0]; wR[1] = sw[1];
        }
        // ---- compute current buffer ----
        const int g   = (lane >> 4) << 3;
        const int kb0 = (lane >> 4) * 8;
        #pragma unroll
        for (int mt = 0; mt < 2; ++mt) {
            int frow = wave * 32 + mt * 16 + (lane & 15);
            // A fragment (16x32): halves 0-7 -> K=g+i, 8-15 -> K=16+g+i
            h8 lo = *(const h8*)&As[cur][frow * LDA + g];
            h8 hi = *(const h8*)&As[cur][frow * LDA + 16 + g];
            v16h a;
            #pragma unroll
            for (int i = 0; i < 8; ++i) { a[i] = lo[i]; a[i + 8] = hi[i]; }
            #pragma unroll
            for (int nt = 0; nt < 4; ++nt) {
                // B fragment (32x16): col = lane&15; halves 0-7 -> K=kb0+i, 8-15 -> K=16+kb0+i
                int n = nt * 16 + (lane & 15);
                h8 b0 = *(const h8*)&Bt[cur][n * LDA + kb0];
                h8 b1 = *(const h8*)&Bt[cur][n * LDA + kb0 + 16];
                v16h b;
                #pragma unroll
                for (int i = 0; i < 8; ++i) { b[i] = b0[i]; b[i + 8] = b1[i]; }
                acc[mt][nt] = __builtin_amdgcn_wmma_f32_16x16x32_f16(
                    false, a, false, b, (short)0, acc[mt][nt], false, false);
            }
        }
        __syncthreads();
        if (more) {
            int nb = cur ^ 1;
            h8* da = (h8*)&As[nb][tid * LDA];
            #pragma unroll
            for (int i = 0; i < 4; ++i) da[i] = aR[i];
            #pragma unroll
            for (int j = 0; j < 4; ++j) Bt[nb][(bn0 + j) * LDA + bk]     = (f16)wR[0][j];
            #pragma unroll
            for (int j = 0; j < 4; ++j) Bt[nb][(bn0 + 4 + j) * LDA + bk] = (f16)wR[1][j];
        }
        __syncthreads();
        cur ^= 1;
    }

    // Epilogue. C layout: element (M,N) = ((lane>>4)*8 + r, lane&15) in VGPR r.
    int nCol0 = nBase + (lane & 15);
    #pragma unroll
    for (int mt = 0; mt < 2; ++mt) {
        int mRow = mBase + wave * 32 + mt * 16 + ((lane >> 4) << 3);
        #pragma unroll
        for (int nt = 0; nt < 4; ++nt) {
            int c = nCol0 + nt * 16;
            #pragma unroll
            for (int r = 0; r < 8; ++r) {
                int m   = mRow + r;
                float v = acc[mt][nt][r] + bias[c];
                if (mode == MODE_QKV) {
                    int which = c / DIM;
                    int rem   = c - which * DIM;
                    int head  = rem >> 6;
                    int d     = rem & 63;
                    int b     = m >> 12, tok = m & 4095;
                    size_t off = ((size_t)(b * HEADS + head) * NTOK + tok) * HD + d;
                    if (which == 0)      outQ[off] = (f16)(v * 0.125f); // 1/sqrt(64)
                    else if (which == 1) outK[off] = (f16)v;
                    else                 outV[off] = (f16)v;
                } else if (mode == MODE_PROJ) {
                    size_t off = (size_t)m * Nw + c;
                    outF[off] = v + resid[off];                // attn residual
                } else if (mode == MODE_FC1) {
                    float g1 = 0.5f * v  * (1.0f + erff(v  * 0.70710678f));
                    float g2 = 0.5f * g1 * (1.0f + erff(g1 * 0.70710678f));
                    outH[(size_t)m * Nw + c] = (f16)g2;        // gelu(gelu(.))
                } else { // MODE_FC2
                    outF[(size_t)m * Nw + c] = v;
                }
            }
        }
    }
}

// ---------------------------------------------------------------------------
// 5) Flash attention, one block per (b, h, 64-query tile); 4 waves.
//    Edge mask enters as additive key bias (0 / -1e9) -> identical math to
//    the reference's masked softmax. Q is pre-scaled by 1/sqrt(hd).
// ---------------------------------------------------------------------------
__global__ __launch_bounds__(128)
void flash_attn_kernel(const f16* __restrict__ Qg, const f16* __restrict__ Kg,
                       const f16* __restrict__ Vg, const float* __restrict__ kbias,
                       f16* __restrict__ Og) {
    __shared__ __align__(16) f16 Qs[64 * LDF];
    __shared__ __align__(16) f16 Ks[64 * LDF];     // [key][d] == transposed-B
    __shared__ __align__(16) f16 Vt[64 * LDF];     // [d][key] (transposed V)
    __shared__ __align__(16) f16 Pw[4][16][LDF];   // per-wave P restage
    __shared__ float kb[64];
    const int b = blockIdx.z, h = blockIdx.y, q0 = blockIdx.x * 64;
    const int tid = threadIdx.x, wave = tid >> 5, lane = tid & 31;
    const size_t headBase = (size_t)(b * HEADS + h) * NTOK * HD;

    // load Q tile 64x64 (vectorized 16B)
    {
        int row = tid >> 1, cb = (tid & 1) * 32;
        const h8* src = (const h8*)(Qg + headBase + (size_t)(q0 + row) * HD + cb);
        h8* dst = (h8*)&Qs[row * LDF + cb];
        #pragma unroll
        for (int i = 0; i < 4; ++i) dst[i] = src[i];
    }

    v8f o[4] = {};
    float m_old[8], lsum[8];
    #pragma unroll
    for (int r = 0; r < 8; ++r) { m_old[r] = -3.0e38f; lsum[r] = 0.f; }

    for (int kt = 0; kt < NTOK / 64; ++kt) {
        __syncthreads();                    // prior iter done reading Ks/Vt/Pw
        {
            int row = tid >> 1, cb = (tid & 1) * 32;
            const h8* sk = (const h8*)(Kg + headBase + (size_t)(kt * 64 + row) * HD + cb);
            h8* dk = (h8*)&Ks[row * LDF + cb];
            #pragma unroll
            for (int i = 0; i < 4; ++i) dk[i] = sk[i];
            const h8* sv = (const h8*)(Vg + headBase + (size_t)(kt * 64 + row) * HD + cb);
            #pragma unroll
            for (int i = 0; i < 4; ++i) {
                h8 vv = sv[i];
                #pragma unroll
                for (int j = 0; j < 8; ++j)
                    Vt[(cb + i * 8 + j) * LDF + row] = vv[j];
            }
            if (tid < 64) kb[tid] = kbias[b * NTOK + kt * 64 + tid];
        }
        __syncthreads();

        // S = Q Kt  (16 q-rows x 64 keys per wave): 2 K-steps x 4 n-tiles
        v8f s[4] = {};
        const int g   = (lane >> 4) << 3;
        #pragma unroll
        for (int ks = 0; ks < 2; ++ks) {
            int row = wave * 16 + (lane & 15);
            h8 lo = *(const h8*)&Qs[row * LDF + ks * 32 + g];
            h8 hi = *(const h8*)&Qs[row * LDF + ks * 32 + 16 + g];
            v16h a;
            #pragma unroll
            for (int i = 0; i < 8; ++i) { a[i] = lo[i]; a[i + 8] = hi[i]; }
            int kb2 = ks * 32 + (lane >> 4) * 8;
            #pragma unroll
            for (int nt = 0; nt < 4; ++nt) {
                int n = nt * 16 + (lane & 15);
                h8 b0 = *(const h8*)&Ks[n * LDF + kb2];
                h8 b1 = *(const h8*)&Ks[n * LDF + kb2 + 16];
                v16h bf;
                #pragma unroll
                for (int i = 0; i < 8; ++i) { bf[i] = b0[i]; bf[i + 8] = b1[i]; }
                s[nt] = __builtin_amdgcn_wmma_f32_16x16x32_f16(
                    false, a, false, bf, (short)0, s[nt], false, false);
            }
        }

        // online softmax (row M = (lane>>4)*8 + r, spread over 16 lanes)
        float kbv[4];
        #pragma unroll
        for (int nt = 0; nt < 4; ++nt) kbv[nt] = kb[nt * 16 + (lane & 15)];
        float m_new[8], alpha[8];
        #pragma unroll
        for (int r = 0; r < 8; ++r) {
            float mx = m_old[r];
            #pragma unroll
            for (int nt = 0; nt < 4; ++nt) mx = fmaxf(mx, s[nt][r] + kbv[nt]);
            #pragma unroll
            for (int msk = 1; msk < 16; msk <<= 1)
                mx = fmaxf(mx, __shfl_xor(mx, msk, 32));
            m_new[r] = mx;
            alpha[r] = expf(m_old[r] - mx);
        }
        #pragma unroll
        for (int r = 0; r < 8; ++r) {
            float rsum = 0.f;
            #pragma unroll
            for (int nt = 0; nt < 4; ++nt) {
                float p = expf(s[nt][r] + kbv[nt] - m_new[r]);
                s[nt][r] = p;
                rsum += p;
            }
            #pragma unroll
            for (int msk = 1; msk < 16; msk <<= 1)
                rsum += __shfl_xor(rsum, msk, 32);
            lsum[r] = lsum[r] * alpha[r] + rsum;
            #pragma unroll
            for (int nt = 0; nt < 4; ++nt) o[nt][r] *= alpha[r];
            m_old[r] = m_new[r];
        }

        // restage P (C layout) into A layout via per-wave LDS
        #pragma unroll
        for (int nt = 0; nt < 4; ++nt)
            #pragma unroll
            for (int r = 0; r < 8; ++r)
                Pw[wave][((lane >> 4) << 3) + r][nt * 16 + (lane & 15)] =
                    (f16)s[nt][r];
        __syncthreads();

        // O += P V  (K = 64 keys -> 2 K-steps; 4 d-tiles)
        #pragma unroll
        for (int ks = 0; ks < 2; ++ks) {
            int row = lane & 15;
            h8 lo = *(const h8*)&Pw[wave][row][ks * 32 + g];
            h8 hi = *(const h8*)&Pw[wave][row][ks * 32 + 16 + g];
            v16h a;
            #pragma unroll
            for (int i = 0; i < 8; ++i) { a[i] = lo[i]; a[i + 8] = hi[i]; }
            int kb2 = ks * 32 + (lane >> 4) * 8;
            #pragma unroll
            for (int nt = 0; nt < 4; ++nt) {
                int n = nt * 16 + (lane & 15);
                h8 b0 = *(const h8*)&Vt[n * LDF + kb2];
                h8 b1 = *(const h8*)&Vt[n * LDF + kb2 + 16];
                v16h bf;
                #pragma unroll
                for (int i = 0; i < 8; ++i) { bf[i] = b0[i]; bf[i + 8] = b1[i]; }
                o[nt] = __builtin_amdgcn_wmma_f32_16x16x32_f16(
                    false, a, false, bf, (short)0, o[nt], false, false);
            }
        }
    }

    // epilogue: O /= l, scatter to [B*N, DIM] f16 (proj GEMM input)
    int qrow = q0 + wave * 16 + ((lane >> 4) << 3);
    int tokBase = b * NTOK;
    #pragma unroll
    for (int r = 0; r < 8; ++r) {
        float inv = 1.0f / lsum[r];
        #pragma unroll
        for (int nt = 0; nt < 4; ++nt) {
            int d = h * HD + nt * 16 + (lane & 15);
            Og[(size_t)(tokBase + qrow + r) * DIM + d] = (f16)(o[nt][r] * inv);
        }
    }
}

// ---------------------------------------------------------------------------
// 6) masked mean pool over easy tokens + ECA conv(k=5, SAME) + sigmoid gate
// ---------------------------------------------------------------------------
__global__ __launch_bounds__(768)
void pool_gate_kernel(const float* __restrict__ h2, const float* __restrict__ edgeF,
                      const float* __restrict__ ecnt, const float* __restrict__ eca_w,
                      float* __restrict__ gate) {
    int b = blockIdx.x, c = threadIdx.x;
    __shared__ float sp[DIM + 4];
    float s = 0.f;
    const float* hp = h2 + (size_t)b * NTOK * DIM;
    const float* ep = edgeF + b * NTOK;
    for (int n = 0; n < NTOK; ++n)
        s += hp[(size_t)n * DIM + c] * (1.0f - ep[n]);
    sp[c + 2] = s / ecnt[b];
    if (c < 2) { sp[c] = 0.f; sp[DIM + 2 + c] = 0.f; }
    __syncthreads();
    float g = eca_w[0] * sp[c]     + eca_w[1] * sp[c + 1] + eca_w[2] * sp[c + 2]
            + eca_w[3] * sp[c + 3] + eca_w[4] * sp[c + 4];
    gate[b * DIM + c] = 1.0f / (1.0f + expf(-g));
}

// ---------------------------------------------------------------------------
// 7) scatter-combine: edge ? attn_out : x + h2 * gate
// ---------------------------------------------------------------------------
__global__ __launch_bounds__(256)
void combine_kernel(const float* __restrict__ x, const float* __restrict__ attn,
                    const float* __restrict__ h2, const float* __restrict__ edgeF,
                    const float* __restrict__ gate, float* __restrict__ out) {
    size_t idx = (size_t)blockIdx.x * 256 + threadIdx.x;
    int c    = (int)(idx % DIM);
    size_t tok = idx / DIM;
    int b    = (int)(tok >> 12);
    float e  = edgeF[tok];
    float lt = x[idx] + h2[idx] * gate[b * DIM + c];
    out[idx] = (e > 0.5f) ? attn[idx] : lt;
}

// ---------------------------------------------------------------------------
extern "C" void kernel_launch(void* const* d_in, const int* in_sizes, int n_in,
                              void* d_out, int out_size, void* d_ws, size_t ws_size,
                              hipStream_t stream) {
    const float* x      = (const float*)d_in[0];
    const float* ah     = (const float*)d_in[1];
    const float* qkv_w  = (const float*)d_in[2];
    const float* qkv_b  = (const float*)d_in[3];
    const float* proj_w = (const float*)d_in[4];
    const float* proj_b = (const float*)d_in[5];
    const float* ln_g   = (const float*)d_in[6];
    const float* ln_b   = (const float*)d_in[7];
    const float* fc1_w  = (const float*)d_in[8];
    const float* fc1_b  = (const float*)d_in[9];
    const float* fc2_w  = (const float*)d_in[10];
    const float* fc2_b  = (const float*)d_in[11];
    const float* eca_w  = (const float*)d_in[12];
    float* out = (float*)d_out;

    char* ws = (char*)d_ws;
    size_t off = 0;
    auto alloc = [&](size_t bytes) -> void* {
        void* p = ws + off;
        off = (off + bytes + 255) & ~(size_t)255;
        return p;
    };
    f16*   x16   = (f16*)  alloc((size_t)MTOT * DIM * 2);
    f16*   xn16  = (f16*)  alloc((size_t)MTOT * DIM * 2);
    f16*   q16   = (f16*)  alloc((size_t)MTOT * DIM * 2);
    f16*   k16   = (f16*)  alloc((size_t)MTOT * DIM * 2);
    f16*   v16   = (f16*)  alloc((size_t)MTOT * DIM * 2);
    f16*   o16   = (f16*)  alloc((size_t)MTOT * DIM * 2);
    f16*   h1    = (f16*)  alloc((size_t)MTOT * HIDDEN * 2);
    float* attn  = (float*)alloc((size_t)MTOT * DIM * 4);
    float* h2    = (float*)alloc((size_t)MTOT * DIM * 4);
    float* edgeF = (float*)alloc((size_t)MTOT * 4);
    float* kbias = (float*)alloc((size_t)MTOT * 4);
    float* gate  = (float*)alloc((size_t)BATCH * DIM * 4);
    float* ecnt  = (float*)alloc(BATCH * 4);
    int*   cnt   = (int*)  alloc(BATCH * 4);

    // 0-2: edge routing mask
    init_counts_kernel<<<1, 32, 0, stream>>>(cnt);
    edge_mask_kernel<<<MTOT / 256, 256, 0, stream>>>(ah, edgeF, cnt);
    edge_finalize_kernel<<<MTOT / 256, 256, 0, stream>>>(edgeF, kbias, cnt, ecnt);

    // 3: LN + f16 conversion
    layernorm_kernel<<<MTOT, 256, 0, stream>>>(x, ln_g, ln_b, x16, xn16);

    // 4: QKV projection (q pre-scaled)
    gemm_wmma_kernel<<<dim3(3 * DIM / 64, MTOT / 256), 256, 0, stream>>>(
        x16, qkv_w, qkv_b, DIM, 3 * DIM, MODE_QKV,
        q16, k16, v16, nullptr, nullptr, nullptr);

    // 5: masked flash attention
    flash_attn_kernel<<<dim3(NTOK / 64, HEADS, BATCH), 128, 0, stream>>>(
        q16, k16, v16, kbias, o16);

    // 6: output projection + residual
    gemm_wmma_kernel<<<dim3(DIM / 64, MTOT / 256), 256, 0, stream>>>(
        o16, proj_w, proj_b, DIM, DIM, MODE_PROJ,
        nullptr, nullptr, nullptr, x, attn, nullptr);

    // 7: fc1 + double GELU
    gemm_wmma_kernel<<<dim3(HIDDEN / 64, MTOT / 256), 256, 0, stream>>>(
        xn16, fc1_w, fc1_b, DIM, HIDDEN, MODE_FC1,
        nullptr, nullptr, nullptr, nullptr, nullptr, h1);

    // 8: fc2
    gemm_wmma_kernel<<<dim3(DIM / 64, MTOT / 256), 256, 0, stream>>>(
        h1, fc2_w, fc2_b, HIDDEN, DIM, MODE_FC2,
        nullptr, nullptr, nullptr, nullptr, h2, nullptr);

    // 9: ECA pool + gate, 10: combine
    pool_gate_kernel<<<BATCH, DIM, 0, stream>>>(h2, edgeF, ecnt, eca_w, gate);
    combine_kernel<<<(int)(((size_t)MTOT * DIM) / 256), 256, 0, stream>>>(
        x, attn, h2, edgeF, gate, out);
}
